// Attention_326417514823
// MI455X (gfx1250) — matrix operand order repaired and checked by reading, not compiled, not run on hardware
//
#include <hip/hip_runtime.h>
#include <hip/hip_bf16.h>

#define NB 8
#define NN 2048
#define ND 256
#define NEG_INF (-1e9f)

// Scheduler fence: nothing may be reordered across it. Keeps the software
// pipeline's load phase physically above the MMA phase.
#define SCHED_FENCE() __builtin_amdgcn_sched_barrier(0)

typedef __attribute__((ext_vector_type(16))) __bf16        v16bf;
typedef __attribute__((ext_vector_type(8)))  float         v8f;
typedef __attribute__((ext_vector_type(4)))  unsigned int  u32x4;

union FragBF {
    u32x4          q[2];
    v16bf          v;
    unsigned short u[16];
};

__device__ __forceinline__ unsigned short f2bf(float x) {
    unsigned int u = __float_as_uint(x);
    u += 0x7FFFu + ((u >> 16) & 1u);   // round to nearest even
    return (unsigned short)(u >> 16);
}

// --- fragment loaders -------------------------------------------------------
__device__ __forceinline__ void load_a2(FragBF* a, const unsigned short* Ab,
                                        int m0, int arow, int d0, int asel) {
#pragma unroll
    for (int i = 0; i < 2; ++i) {
        const unsigned short* pa = Ab + (m0 + 16 * i + arow) * ND + d0 + asel;
        a[i].q[0] = *reinterpret_cast<const u32x4*>(pa);
        a[i].q[1] = *reinterpret_cast<const u32x4*>(pa + 16);
    }
}
__device__ __forceinline__ void load_b4(FragBF* bf, const unsigned short* prow, int ld) {
#pragma unroll
    for (int j = 0; j < 4; ++j) {
        bf[j].q[0] = *reinterpret_cast<const u32x4*>(prow + 16 * j * ld);
        bf[j].q[1] = *reinterpret_cast<const u32x4*>(prow + 16 * j * ld + 16);
    }
}
__device__ __forceinline__ void mma8(v8f acc[2][4], const FragBF* a, const FragBF* bf) {
#pragma unroll
    for (int i = 0; i < 2; ++i)
#pragma unroll
        for (int j = 0; j < 4; ++j)
            acc[i][j] = __builtin_amdgcn_wmma_f32_16x16x32_bf16(
                false, a[i].v, false, bf[j].v, (short)0, acc[i][j], false, false);
}
__device__ __forceinline__ void mma4(v8f* acc, v16bf av, const FragBF* bf) {
#pragma unroll
    for (int j = 0; j < 4; ++j)
        acc[j] = __builtin_amdgcn_wmma_f32_16x16x32_bf16(
            false, av, false, bf[j].v, (short)0, acc[j], false, false);
}

// ---------------------------------------------------------------------------
// Kernel 0: Q * (1/16) -> bf16, K -> bf16 (b,n,d), V -> bf16 transposed (b,d,n).
// ---------------------------------------------------------------------------
__global__ void __launch_bounds__(256)
prep_kernel(const float* __restrict__ key, const float* __restrict__ query,
            const float* __restrict__ value,
            unsigned short* __restrict__ qb, unsigned short* __restrict__ ktb,
            unsigned short* __restrict__ vb) {
    int i = blockIdx.x * 256 + threadIdx.x;          // exact: NB*NN*ND / 256
    qb[i] = f2bf(query[i] * 0.0625f);

    int b = i / (NN * ND);
    int r = i - b * (NN * ND);
    int n = r / ND;
    int d = r - n * ND;
    ktb[i] = f2bf(key[i]);
    vb[b * (NN * ND) + d * NN + n] = f2bf(value[i]);
}

// ---------------------------------------------------------------------------
// Kernel 1: S = Q K^T. One wave -> 32x64 tile, 64 WMMAs. Double-buffered and
// phase-fenced with sched_barrier so the next block's 12 b128 loads issue as
// one clause before the current block's 8 WMMAs.
// ---------------------------------------------------------------------------
__global__ void __launch_bounds__(256, 1)
gemm1_qk(const unsigned short* __restrict__ qb,
         const unsigned short* __restrict__ ktb,
         const int* __restrict__ mask,
         float* __restrict__ S) {
    const int lane = threadIdx.x & 31;
    const int wave = (blockIdx.x << 3) | (threadIdx.x >> 5);
    const int tilesPerBatch = (NN / 32) * (NN / 64);        // 64*32 = 2048
    const int b  = wave / tilesPerBatch;
    const int t  = wave - b * tilesPerBatch;
    const int qt = t / (NN / 64);                           // 0..63
    const int kt = t - qt * (NN / 64);                      // 0..31
    const int m0 = qt * 32, n0 = kt * 64;

    const unsigned short* Ab = qb  + b * (NN * ND);
    const unsigned short* Bb = ktb + b * (ND * NN);

    const int arow = lane & 15;
    const int asel = (lane & 16) ? 8 : 0;       // A K-segment select (ISA layout)

    v8f acc[2][4] = {};
    FragBF a0[2], b0[4], a1[2], b1[4];
    load_a2(a0, Ab, m0, arow, 0, asel);
    load_b4(b0, Bb + (n0 + arow) * ND + asel, ND);

#pragma unroll
    for (int d0 = 0; d0 < ND; d0 += 64) {
        load_a2(a1, Ab, m0, arow, d0 + 32, asel);
        load_b4(b1, Bb + (n0 + arow) * ND + d0 + 32 + asel, ND);
        SCHED_FENCE();
        mma8(acc, a0, b0);
        SCHED_FENCE();
        if (d0 + 64 < ND) {
            load_a2(a0, Ab, m0, arow, d0 + 64, asel);
            load_b4(b0, Bb + (n0 + arow) * ND + d0 + 64 + asel, ND);
        }
        SCHED_FENCE();
        mma8(acc, a1, b1);
        SCHED_FENCE();
    }

    const int hi = (lane & 16) ? 8 : 0;
    const int nl = lane & 15;
    float* Sb = S + (size_t)b * NN * NN;
#pragma unroll
    for (int i = 0; i < 2; ++i) {
#pragma unroll
        for (int r = 0; r < 8; ++r) {
            int q = m0 + 16 * i + hi + r;
            int mk = mask[b * NN + q];
#pragma unroll
            for (int j = 0; j < 4; ++j) {
                float val = mk ? acc[i][j][r] : NEG_INF;
                Sb[(size_t)q * NN + n0 + 16 * j + nl] = val;
            }
        }
    }
}

// ---------------------------------------------------------------------------
// Kernel 2a: partial column stats over q-chunks of 256 (8 chunks per column).
// ---------------------------------------------------------------------------
__global__ void __launch_bounds__(256)
colstats_part(const float* __restrict__ S,
              float* __restrict__ pmax, float* __restrict__ psum) {
    int idx = blockIdx.x * 256 + threadIdx.x;   // (b, chunk, k)
    int b = idx / (8 * NN);
    int r = idx - b * (8 * NN);
    int c = r / NN;
    int k = r - c * NN;
    const float* p = S + (size_t)b * NN * NN + (size_t)(c * 256) * NN + k;

    float m = -3.4e38f;
#pragma unroll 8
    for (int q = 0; q < 256; ++q) m = fmaxf(m, p[(size_t)q * NN]);
    float s = 0.0f;
#pragma unroll 8
    for (int q = 0; q < 256; ++q) s += __expf(p[(size_t)q * NN] - m);

    pmax[idx] = m;
    psum[idx] = s;
}

// ---------------------------------------------------------------------------
// Kernel 2b: merge 8 partial (max,sum) pairs -> colmax, colrsum = 1/sum.
// ---------------------------------------------------------------------------
__global__ void __launch_bounds__(256)
colstats_merge(const float* __restrict__ pmax, const float* __restrict__ psum,
               float* __restrict__ colmax, float* __restrict__ colrsum) {
    int idx = blockIdx.x * 256 + threadIdx.x;   // b*NN + k
    int b = idx / NN;
    int k = idx - b * NN;
    const float* pm = pmax + (size_t)b * (8 * NN) + k;
    const float* ps = psum + (size_t)b * (8 * NN) + k;

    float m = -3.4e38f;
#pragma unroll
    for (int c = 0; c < 8; ++c) m = fmaxf(m, pm[c * NN]);
    float s = 0.0f;
#pragma unroll
    for (int c = 0; c < 8; ++c) s += ps[c * NN] * __expf(pm[c * NN] - m);

    colmax[idx]  = m;
    colrsum[idx] = 1.0f / s;
}

// --- gemm2 helpers ----------------------------------------------------------
__device__ __forceinline__ void load_s16(float* sv, const float* ps) {
#pragma unroll
    for (int j = 0; j < 16; ++j) sv[j] = ps[(j < 8) ? j : (j + 8)];
}
__device__ __forceinline__ v16bf make_afrag(const float* sv, const float* pm,
                                            const float* pr, int mk) {
    FragBF a;
#pragma unroll
    for (int j = 0; j < 16; ++j) {
        int off = (j < 8) ? j : (j + 8);
        float e = __expf(sv[j] - pm[off]) * pr[off];
        a.u[j] = mk ? f2bf(e) : (unsigned short)0;
    }
    return a.v;
}

// ---------------------------------------------------------------------------
// Kernel 3: O = A V, A = mask(q)*exp(S-colmax)*colrsum built on the fly in
// the bf16 A-operand layout. One wave -> 16x64 tile; phase-fenced double
// buffering; exp (TRANS) co-executes with WMMA inside the compute phase.
// ---------------------------------------------------------------------------
__global__ void __launch_bounds__(256, 1)
gemm2_av(const float* __restrict__ S,
         const unsigned short* __restrict__ vb,
         const float* __restrict__ colmax, const float* __restrict__ colrsum,
         const int* __restrict__ mask,
         float* __restrict__ out) {
    const int lane = threadIdx.x & 31;
    const int wave = (blockIdx.x << 3) | (threadIdx.x >> 5);
    const int tilesPerBatch = (NN / 16) * (ND / 64);        // 128*4 = 512
    const int b  = wave / tilesPerBatch;
    const int t  = wave - b * tilesPerBatch;
    const int qt = t / (ND / 64);
    const int dg = t - qt * (ND / 64);
    const int m0 = qt * 16, n0 = dg * 64;

    const float*          Sb = S  + (size_t)b * NN * NN;
    const unsigned short* Vb = vb + b * (NN * ND);
    const float*          cm = colmax  + b * NN;
    const float*          cr = colrsum + b * NN;

    const int q    = m0 + (lane & 15);
    const int ksel = (lane & 16) ? 8 : 0;
    const int mk   = mask[b * NN + q];
    const float* Srow = Sb + (size_t)q * NN;

    v8f acc[4] = {};
    float  sv0[16], sv1[16];
    FragBF b0[4], b1[4];
    load_s16(sv0, Srow + ksel);
    load_b4(b0, Vb + (n0 + (lane & 15)) * NN + ksel, NN);

    for (int k0 = 0; k0 < NN; k0 += 64) {
        __builtin_prefetch(Srow + k0 + 128 + ksel, 0, 1);   // stream next S blocks
        load_s16(sv1, Srow + k0 + 32 + ksel);
        load_b4(b1, Vb + (n0 + (lane & 15)) * NN + k0 + 32 + ksel, NN);
        SCHED_FENCE();
        mma4(acc, make_afrag(sv0, cm + k0 + ksel, cr + k0 + ksel, mk), b0);
        SCHED_FENCE();
        if (k0 + 64 < NN) {
            load_s16(sv0, Srow + k0 + 64 + ksel);
            load_b4(b0, Vb + (n0 + (lane & 15)) * NN + k0 + 64 + ksel, NN);
        }
        SCHED_FENCE();
        mma4(acc, make_afrag(sv1, cm + k0 + 32 + ksel, cr + k0 + 32 + ksel, mk), b1);
        SCHED_FENCE();
    }

    const int hi = (lane & 16) ? 8 : 0;
    const int nl = lane & 15;
#pragma unroll
    for (int j = 0; j < 4; ++j)
#pragma unroll
        for (int r = 0; r < 8; ++r)
            out[((size_t)b * NN + m0 + hi + r) * ND + n0 + 16 * j + nl] = acc[j][r];
}

// ---------------------------------------------------------------------------
// Workspace layout (bytes):
//   [0,8M) qb | [8M,16M) ktb | [16M,24M) vb | [24M,152M) S (fits 192MB L2)
//   [152M,+64K) colmax | [+64K,+128K) colrsum
//   [153M,+512K) pmax  | [+512K,+1M)  psum          Total ~154 MB.
// ---------------------------------------------------------------------------
extern "C" void kernel_launch(void* const* d_in, const int* in_sizes, int n_in,
                              void* d_out, int out_size, void* d_ws, size_t ws_size,
                              hipStream_t stream) {
    (void)in_sizes; (void)n_in; (void)out_size; (void)ws_size;
    const float* key   = (const float*)d_in[0];
    const float* query = (const float*)d_in[1];
    const float* value = (const float*)d_in[2];
    const int*   mask  = (const int*)d_in[3];
    float*       out   = (float*)d_out;

    char* ws = (char*)d_ws;
    const size_t MB = 1024ull * 1024ull;
    unsigned short* qb      = (unsigned short*)(ws);
    unsigned short* ktb     = (unsigned short*)(ws + 8 * MB);
    unsigned short* vb      = (unsigned short*)(ws + 16 * MB);
    float*          S       = (float*)(ws + 24 * MB);
    float*          colmax  = (float*)(ws + 152 * MB);
    float*          colrsum = (float*)(ws + 152 * MB + 64 * 1024);
    float*          pmax    = (float*)(ws + 153 * MB);
    float*          psum    = (float*)(ws + 153 * MB + 512 * 1024);

    prep_kernel<<<(NB * NN * ND) / 256, 256, 0, stream>>>(key, query, value, qb, ktb, vb);
    gemm1_qk<<<(NB * (NN / 32) * (NN / 64)) / 8, 256, 0, stream>>>(qb, ktb, mask, S);
    colstats_part<<<(NB * 8 * NN) / 256, 256, 0, stream>>>(S, pmax, psum);
    colstats_merge<<<(NB * NN) / 256, 256, 0, stream>>>(pmax, psum, colmax, colrsum);
    gemm2_av<<<(NB * (NN / 16) * (ND / 64)) / 8, 256, 0, stream>>>(S, vb, colmax, colrsum, mask, out);
}
